// VectorQuantizer_36309653520635
// MI455X (gfx1250) — matrix operand order repaired and checked by reading, not compiled, not run on hardware
//
#include <hip/hip_runtime.h>

typedef __attribute__((ext_vector_type(16))) __bf16          v16bf;
typedef __attribute__((ext_vector_type(16))) unsigned short  v16u;
typedef __attribute__((ext_vector_type(8)))  float           v8f;

#define NTOK   32768          // 32*1024 tokens
#define DDIM   256
#define KCODE  8192
#define NTILES (KCODE / 16)   // 512 code tiles of 16
#define CHUNKS (DDIM / 32)    // 8 k-chunks of 32 per dot product

// workspace layout (bytes)
#define WS_ENORM   0                  // 8192 f32  (32KB)
#define WS_COUNTS  (32*1024)          // 8192 f32  (32KB)
#define WS_SSE     (64*1024)          // 1 f32
#define WS_BFRAG   (1u<<20)           // bf16 hi/lo B fragments, 8MB

__device__ __forceinline__ unsigned short f32_to_bf16_rne(float f) {
    unsigned u = __float_as_uint(f);
    unsigned r = u + 0x7FFFu + ((u >> 16) & 1u);
    return (unsigned short)(r >> 16);
}
__device__ __forceinline__ float bf16_bits_to_f32(unsigned short s) {
    return __uint_as_float(((unsigned)s) << 16);
}

// ---------------------------------------------------------------------------
// Kernel 1: convert codebook to WMMA B-fragment layout (bf16 hi + lo),
// compute ||e||^2 per code, zero counts / sse accumulators.
// One block per 16-code tile. B frag slot = (nt*8 + c)*2 + h ; per slot
// 32 lanes x 16 halfs. Lane holds column n = lane&15, K = (lane>>4)*16 + j.
// Linear layout: frag pair f = nt*8+c at byte offset f*2048 (+1024 for lo).
// ---------------------------------------------------------------------------
__global__ void vq_preproc(const float* __restrict__ emb,
                           unsigned short* __restrict__ bfr,
                           float* __restrict__ enorm,
                           float* __restrict__ counts,
                           float* __restrict__ sse) {
    const int nt  = blockIdx.x;
    const int tid = threadIdx.x;

    if (tid < 16) counts[nt * 16 + tid] = 0.0f;
    if (nt == 0 && tid == 0) sse[0] = 0.0f;

    // ---- fragment conversion: thread -> (chunk c, lane) ----
    const int c    = tid >> 5;
    const int lane = tid & 31;
    const int col  = lane & 15;
    const int d0   = c * 32 + ((lane >> 4) * 16);   // 16 contiguous K values

    const float* pe = emb + (size_t)(nt * 16 + col) * DDIM + d0;
    v16u uh, ul;
#pragma unroll
    for (int j = 0; j < 16; ++j) {
        float x = pe[j];
        unsigned short h = f32_to_bf16_rne(x);
        float residual = x - bf16_bits_to_f32(h);
        uh[j] = h;
        ul[j] = f32_to_bf16_rne(residual);
    }
    const size_t fp = (size_t)(nt * CHUNKS + c) * 1024;   // ushort units
    *(v16u*)(bfr + fp + (size_t)lane * 16)       = uh;
    *(v16u*)(bfr + fp + 512 + (size_t)lane * 16) = ul;

    // ---- ||e||^2 : tid = part*16 + code, 16 partials of 16 elems ----
    __shared__ float sm[256];
    const int code = tid & 15;
    const int part = tid >> 4;
    const float* pn = emb + (size_t)(nt * 16 + code) * DDIM + part * 16;
    float s = 0.0f;
#pragma unroll
    for (int j = 0; j < 16; ++j) { float v = pn[j]; s += v * v; }
    sm[tid] = s;
    __syncthreads();
    if (part == 0) {
        float t = 0.0f;
#pragma unroll
        for (int p = 0; p < 16; ++p) t += sm[p * 16 + code];
        enorm[nt * 16 + code] = t;
    }
}

// ---------------------------------------------------------------------------
// Kernel 2: fused GEMM (bf16 hi/lo WMMA) + argmin + gather + loss partials.
// 8 waves/block; each wave owns TWO 16-row tiles (32 rows) so every B
// fragment pair feeds 6 WMMAs. B loads software-pipelined one chunk ahead.
// ---------------------------------------------------------------------------
__global__ void __launch_bounds__(256)
vq_main(const float* __restrict__ x,
        const float* __restrict__ emb,
        const unsigned short* __restrict__ bfr,
        const float* __restrict__ enorm,
        float* __restrict__ counts,
        float* __restrict__ sse,
        float* __restrict__ out_q,
        float* __restrict__ out_idx) {
    const int wave = threadIdx.x >> 5;
    const int lane = threadIdx.x & 31;
    const int rowBase = blockIdx.x * 256 + wave * 32;

    // ---- build A fragments (bf16 hi/lo) for both row tiles, in registers ----
    // A layout: lane holds row M = lane&15 ; 16 contiguous K at koff,
    // koff = (lane>>4)*16.
    const int koff = (lane >> 4) * 16;

    v16bf Ah0[CHUNKS], Al0[CHUNKS], Ah1[CHUNKS], Al1[CHUNKS];
#pragma unroll
    for (int tle = 0; tle < 2; ++tle) {
        const int rowA = rowBase + tle * 16 + (lane & 15);
#pragma unroll
        for (int c = 0; c < CHUNKS; ++c) {
            const float* px = x + (size_t)rowA * DDIM + c * 32 + koff;
            v16u uh, ul;
#pragma unroll
            for (int j = 0; j < 16; ++j) {
                float v = px[j];
                unsigned short h = f32_to_bf16_rne(v);
                uh[j] = h;
                ul[j] = f32_to_bf16_rne(v - bf16_bits_to_f32(h));
            }
            if (tle == 0) {
                Ah0[c] = __builtin_bit_cast(v16bf, uh);
                Al0[c] = __builtin_bit_cast(v16bf, ul);
            } else {
                Ah1[c] = __builtin_bit_cast(v16bf, uh);
                Al1[c] = __builtin_bit_cast(v16bf, ul);
            }
        }
    }

    float minv0[8], minv1[8];
    int   mini0[8], mini1[8];
#pragma unroll
    for (int r = 0; r < 8; ++r) {
        minv0[r] = 3.402823466e38f; mini0[r] = 0;
        minv1[r] = 3.402823466e38f; mini1[r] = 0;
    }

    // ---- software-pipelined B fragment stream: prefetch chunk 0 ----
    const unsigned short* pB = bfr + (size_t)lane * 16;
    v16u bhu = *(const v16u*)(pB);
    v16u blu = *(const v16u*)(pB + 512);
    pB += 1024;

    // ---- sweep all code tiles ----
#pragma unroll 1
    for (int nt = 0; nt < NTILES; ++nt) {
        v8f acc0a = {}, acc1a = {};
        v8f acc0b = {}, acc1b = {};
#pragma unroll
        for (int c = 0; c < CHUNKS; ++c) {
            v16bf Bh = __builtin_bit_cast(v16bf, bhu);
            v16bf Bl = __builtin_bit_cast(v16bf, blu);
            // prefetch next fragment pair (uniform guard only in c==7 copy)
            if ((c != CHUNKS - 1) || (nt != NTILES - 1)) {
                bhu = *(const v16u*)(pB);
                blu = *(const v16u*)(pB + 512);
                pB += 1024;
            }
            acc0a = __builtin_amdgcn_wmma_f32_16x16x32_bf16(
                false, Ah0[c], false, Bh, (short)0, acc0a, false, false);
            acc1a = __builtin_amdgcn_wmma_f32_16x16x32_bf16(
                false, Ah0[c], false, Bl, (short)0, acc1a, false, false);
            acc1a = __builtin_amdgcn_wmma_f32_16x16x32_bf16(
                false, Al0[c], false, Bh, (short)0, acc1a, false, false);
            acc0b = __builtin_amdgcn_wmma_f32_16x16x32_bf16(
                false, Ah1[c], false, Bh, (short)0, acc0b, false, false);
            acc1b = __builtin_amdgcn_wmma_f32_16x16x32_bf16(
                false, Ah1[c], false, Bl, (short)0, acc1b, false, false);
            acc1b = __builtin_amdgcn_wmma_f32_16x16x32_bf16(
                false, Al1[c], false, Bh, (short)0, acc1b, false, false);
        }
        const int colg = nt * 16 + (lane & 15);
        const float en = enorm[colg];
        // C/D layout: VGPR r -> row r (lanes 0-15) / r+8 (lanes 16-31), col = lane&15
#pragma unroll
        for (int r = 0; r < 8; ++r) {
            float d0 = en - 2.0f * (acc0a[r] + acc1a[r]);
            if (d0 < minv0[r]) { minv0[r] = d0; mini0[r] = colg; }
            float d1 = en - 2.0f * (acc0b[r] + acc1b[r]);
            if (d1 < minv1[r]) { minv1[r] = d1; mini1[r] = colg; }
        }
    }

    // ---- cross-lane argmin within each 16-lane half (tie -> lower index) ----
#pragma unroll
    for (int r = 0; r < 8; ++r) {
#pragma unroll
        for (int m = 1; m < 16; m <<= 1) {
            float ov = __shfl_xor(minv0[r], m, 32);
            int   oi = __shfl_xor(mini0[r], m, 32);
            if (ov < minv0[r] || (ov == minv0[r] && oi < mini0[r])) {
                minv0[r] = ov; mini0[r] = oi;
            }
            float pv = __shfl_xor(minv1[r], m, 32);
            int   pi = __shfl_xor(mini1[r], m, 32);
            if (pv < minv1[r] || (pv == minv1[r] && pi < mini1[r])) {
                minv1[r] = pv; mini1[r] = pi;
            }
        }
    }

    // ---- epilogue: per-row gather, quantized write, SSE, counts, indices ----
    float sacc = 0.0f;
#pragma unroll
    for (int t = 0; t < 32; ++t) {
        const int tt  = t & 15;                 // row within tile
        const int src = (tt >> 3) * 16;         // lane holding the reduced value
        const int idx = (t < 16) ? __shfl(mini0[tt & 7], src, 32)
                                 : __shfl(mini1[tt & 7], src, 32);
        const int grow = rowBase + t;
        if (lane == tt) {
            out_idx[grow] = (float)idx;
            atomicAdd(&counts[idx], 1.0f);
        }
        const float* pe  = emb   + (size_t)idx  * DDIM + lane * 8;
        const float* pxr = x     + (size_t)grow * DDIM + lane * 8;
        float*       pq  = out_q + (size_t)grow * DDIM + lane * 8;
#pragma unroll
        for (int j = 0; j < 8; ++j) {
            float e = pe[j];
            float d = e - pxr[j];
            pq[j] = e;
            sacc += d * d;
        }
    }
#pragma unroll
    for (int m = 1; m < 32; m <<= 1) sacc += __shfl_xor(sacc, m, 32);
    if (lane == 0) atomicAdd(sse, sacc);
}

// ---------------------------------------------------------------------------
// Kernel 3: finalize loss + perplexity.
// ---------------------------------------------------------------------------
__global__ void vq_finalize(const float* __restrict__ counts,
                            const float* __restrict__ sse,
                            float* __restrict__ out_loss,
                            float* __restrict__ out_perp) {
    __shared__ float sm[256];
    const int tid = threadIdx.x;
    float s = 0.0f;
    for (int i = tid; i < KCODE; i += 256) {
        float p = counts[i] * (1.0f / (float)NTOK);
        s += p * logf(p + 1e-10f);
    }
    sm[tid] = s;
    __syncthreads();
    for (int off = 128; off > 0; off >>= 1) {
        if (tid < off) sm[tid] += sm[tid + off];
        __syncthreads();
    }
    if (tid == 0) {
        out_loss[0] = 1.25f * sse[0] / (float)(NTOK * DDIM);
        out_perp[0] = expf(-sm[0]);
    }
}

extern "C" void kernel_launch(void* const* d_in, const int* in_sizes, int n_in,
                              void* d_out, int out_size, void* d_ws, size_t ws_size,
                              hipStream_t stream) {
    const float* x   = (const float*)d_in[0];   // [32,1024,256] f32
    const float* emb = (const float*)d_in[1];   // [8192,256]   f32

    char* ws = (char*)d_ws;
    float*          enorm  = (float*)(ws + WS_ENORM);
    float*          cnts   = (float*)(ws + WS_COUNTS);
    float*          sse    = (float*)(ws + WS_SSE);
    unsigned short* bfr    = (unsigned short*)(ws + WS_BFRAG);

    float* out      = (float*)d_out;
    float* out_loss = out;                         // [1]
    float* out_q    = out + 1;                     // [8388608]
    float* out_perp = out + 1 + NTOK * DDIM;       // [1]
    float* out_idx  = out + 2 + NTOK * DDIM;       // [32768]

    vq_preproc<<<NTILES, 256, 0, stream>>>(emb, bfr, enorm, cnts, sse);
    vq_main<<<NTOK / 256, 256, 0, stream>>>(x, emb, bfr, enorm, cnts, sse,
                                            out_q, out_idx);
    vq_finalize<<<1, 256, 0, stream>>>(cnts, sse, out_loss, out_perp);
}
